// EfficientPairedAttention_78941498901119
// MI455X (gfx1250) — matrix operand order repaired and checked by reading, not compiled, not run on hardware
//
#include <hip/hip_runtime.h>

typedef _Float16 f16;
typedef __attribute__((ext_vector_type(16))) _Float16 v16h;
typedef __attribute__((ext_vector_type(8)))  _Float16 h8;
typedef __attribute__((ext_vector_type(4)))  _Float16 h4;
typedef __attribute__((ext_vector_type(8)))  float    v8f;
typedef __attribute__((ext_vector_type(2)))  float    f2;

#define WMMA_F16(a,b,c) __builtin_amdgcn_wmma_f32_16x16x32_f16(false,(a),false,(b),(short)0,(c),false,false)

constexpr int BB   = 4;
constexpr int NTOK = 8192;
constexpr int CCH  = 512;
constexpr int HH   = 8;
constexpr int KP   = 256;
constexpr int DD   = 64;
constexpr int ROWS = BB * NTOK;   // 32768
constexpr int C4   = 4 * CCH;     // 2048
constexpr int NZ   = 8;           // n-slices for partial reductions

// ---------------------------------------------------------------------------
// K-1a: x f32 -> X16 f16 (same [ROWS][512] layout), vectorized
// ---------------------------------------------------------------------------
__global__ __launch_bounds__(256) void k_cvt_x(const float* __restrict__ X,
                                               f16* __restrict__ X16) {
  size_t i = ((size_t)blockIdx.x * 256 + threadIdx.x) * 4;
  float4 v = *(const float4*)(X + i);
  h4 o;
  o[0] = (f16)v.x; o[1] = (f16)v.y; o[2] = (f16)v.z; o[3] = (f16)v.w;
  *(h4*)(X16 + i) = o;
}

// ---------------------------------------------------------------------------
// K-1b: w [512][2048] f32 -> WT16 [2048][512] f16 (k-contiguous rows)
// ---------------------------------------------------------------------------
__global__ __launch_bounds__(256) void k_transpose_w(const float* __restrict__ W,
                                                     f16* __restrict__ WT16) {
  __shared__ f16 tile[64 * 66];
  int k0 = blockIdx.x * 64, c0 = blockIdx.y * 64;
  int tj = threadIdx.x & 63, ti = threadIdx.x >> 6;
  for (int i = ti; i < 64; i += 4)
    tile[tj * 66 + i] = (f16)W[(size_t)(k0 + i) * C4 + (c0 + tj)];
  __syncthreads();
  for (int r = ti; r < 64; r += 4)
    WT16[(size_t)(c0 + r) * CCH + (k0 + tj)] = tile[r * 66 + tj];
}

// ---------------------------------------------------------------------------
// K0: transpose spatial_proj [N,K] f32 -> PT [K,N] f16
// ---------------------------------------------------------------------------
__global__ __launch_bounds__(256) void k_transpose_P(const float* __restrict__ P,
                                                     f16* __restrict__ PT) {
  __shared__ f16 tile[64 * 66];
  int n0 = blockIdx.x * 64, k0 = blockIdx.y * 64;
  int tj = threadIdx.x & 63, ti = threadIdx.x >> 6;
  for (int i = ti; i < 64; i += 4)
    tile[tj * 66 + i] = (f16)P[(size_t)(n0 + i) * KP + (k0 + tj)];
  __syncthreads();
  for (int r = ti; r < 64; r += 4)
    PT[(size_t)(k0 + r) * NTOK + (n0 + tj)] = tile[r * 66 + tj];
}

// ---------------------------------------------------------------------------
// K1: qkvv = x @ w (f16 WMMA, f32 accum), operands loaded per-lane directly
// from X16 / WT16 (both in WMMA-friendly k-contiguous layout; L2-resident).
// C tiles staged in LDS [c][m], then:
//   q_nat, vca_nat : coalesced b64 row stores      [ROWS][512] f16
//   q_t, k_t, vsa_t: async LDS->global b64 stores  [B,H,D,N]   f16
// ---------------------------------------------------------------------------
__global__ __launch_bounds__(256) void k_qkvv(const f16* __restrict__ X16,
                                              const f16* __restrict__ WT16,
                                              f16* __restrict__ q_nat,
                                              f16* __restrict__ vca_nat,
                                              f16* __restrict__ q_t,
                                              f16* __restrict__ k_t,
                                              f16* __restrict__ vsa_t) {
  __shared__ f16 Cs[128 * 132];

  const int wave = threadIdx.x >> 5, lane = threadIdx.x & 31;
  const int lcol = lane & 15, half = lane >> 4;
  const int mw = wave & 3, nw = wave >> 2;   // 4x2 wave grid -> 128x128 macro tile
  const int row0 = blockIdx.x * 128;
  const int c0 = blockIdx.y * 128;
  const int t = c0 >> 9;            // 0:q 1:k 2:vca 3:vsa (tile never crosses)
  const int cT = c0 & 511;

  const f16* arow = X16 + (size_t)(row0 + mw * 32 + lcol) * CCH;
  const f16* brow = WT16 + (size_t)(c0 + nw * 64 + lcol) * CCH;

  v8f acc[2][4] = {};
  for (int ks = 0; ks < CCH; ks += 32) {
    v16h a[2];
#pragma unroll
    for (int mt = 0; mt < 2; ++mt) {
      const f16* p = arow + (size_t)mt * 16 * CCH + ks + half * 8;
      ((h8*)&a[mt])[0] = *(const h8*)p;
      ((h8*)&a[mt])[1] = *(const h8*)(p + 16);
    }
#pragma unroll
    for (int nt = 0; nt < 4; ++nt) {
      const f16* p = brow + (size_t)nt * 16 * CCH + ks + half * 8;
      v16h bb;
      ((h8*)&bb)[0] = *(const h8*)p;
      ((h8*)&bb)[1] = *(const h8*)(p + 16);
#pragma unroll
      for (int mt = 0; mt < 2; ++mt) acc[mt][nt] = WMMA_F16(a[mt], bb, acc[mt][nt]);
    }
  }

  // stage C macro-tile into LDS as [c][m], pitch 132
#pragma unroll
  for (int mt = 0; mt < 2; ++mt)
#pragma unroll
    for (int nt = 0; nt < 4; ++nt) {
      f16* cp = &Cs[(nw * 64 + nt * 16 + lcol) * 132 + mw * 32 + mt * 16 + 8 * half];
#pragma unroll
      for (int j = 0; j < 8; ++j) cp[j] = (f16)acc[mt][nt][j];
    }
  __syncthreads();

  // natural layout: row m needs c-contiguous -> gather 4 c's, coalesced b64 store
  if (t == 0 || t == 2) {
    f16* nat = (t == 0) ? q_nat : vca_nat;
    int cb = lane * 4;
#pragma unroll
    for (int r = 0; r < 16; ++r) {
      int m = wave * 16 + r;
      h4 v;
#pragma unroll
      for (int i = 0; i < 4; ++i) v[i] = Cs[(cb + i) * 132 + m];
      *(h4*)(nat + (size_t)(row0 + m) * CCH + cT + cb) = v;
    }
  }
  // transposed layout: LDS row c is n-contiguous -> async LDS->global copy
  if (t != 2) {
    f16* dst = (t == 0) ? q_t : (t == 1 ? k_t : vsa_t);
    int b = row0 >> 13;
    int nbase = row0 & (NTOK - 1);
#pragma unroll
    for (int r = 0; r < 16; ++r) {
      int c = wave * 16 + r;
      int cch = cT + c;
      int hh = cch >> 6, dd = cch & 63;
      const f16* gp = dst + (((size_t)(b * HH + hh) * DD + dd) * NTOK) + nbase + lane * 4;
      unsigned lds = (unsigned)(size_t)(const void*)&Cs[c * 132 + lane * 4];
      asm volatile("global_store_async_from_lds_b64 %0, %1, off"
                   :: "v"(gp), "v"(lds) : "memory");
    }
    asm volatile("s_wait_asynccnt 0x0" ::: "memory");
  }
}

// ---------------------------------------------------------------------------
// K2: rq/rk[b,h,d] = 1/sqrt(max(sum_n v^2, eps))
// ---------------------------------------------------------------------------
__global__ __launch_bounds__(256) void k_norms(const f16* __restrict__ q_t,
                                               const f16* __restrict__ k_t,
                                               float* __restrict__ rq,
                                               float* __restrict__ rk) {
  __shared__ float sA[256], sB[256];
  size_t base = (size_t)blockIdx.x * NTOK;
  float a = 0.f, c = 0.f;
  for (int i = threadIdx.x; i < NTOK / 8; i += 256) {
    h8 qv = *(const h8*)(q_t + base + (size_t)i * 8);
    h8 kv = *(const h8*)(k_t + base + (size_t)i * 8);
#pragma unroll
    for (int e = 0; e < 8; ++e) {
      float q = (float)qv[e]; a += q * q;
      float k = (float)kv[e]; c += k * k;
    }
  }
  sA[threadIdx.x] = a; sB[threadIdx.x] = c;
  __syncthreads();
  for (int s = 128; s > 0; s >>= 1) {
    if (threadIdx.x < (unsigned)s) {
      sA[threadIdx.x] += sA[threadIdx.x + s];
      sB[threadIdx.x] += sB[threadIdx.x + s];
    }
    __syncthreads();
  }
  if (threadIdx.x == 0) {
    float x = sA[0]; if (x < 1e-6f) x = 1e-6f;
    float y = sB[0]; if (y < 1e-6f) y = 1e-6f;
    rq[blockIdx.x] = 1.0f / sqrtf(x);
    rk[blockIdx.x] = 1.0f / sqrtf(y);
  }
}

// ---------------------------------------------------------------------------
// K3a: Gram partials S_z[b,h,d,e] = sum_{n in slice z} q[d,n] k[e,n]
// ---------------------------------------------------------------------------
__global__ __launch_bounds__(128) void k_gram(const f16* __restrict__ q_t,
                                              const f16* __restrict__ k_t,
                                              float* __restrict__ SgramZ) {
  const int bh = blockIdx.x, z = blockIdx.y;
  const int wave = threadIdx.x >> 5, lane = threadIdx.x & 31;
  const int lcol = lane & 15, half = lane >> 4;
  const size_t tb = (size_t)bh * DD * NTOK;
  v8f acc[4] = {};
  const f16* arow = q_t + tb + (size_t)(wave * 16 + lcol) * NTOK;
  for (int n = z * (NTOK / NZ); n < (z + 1) * (NTOK / NZ); n += 32) {
    v16h a;
    ((h8*)&a)[0] = *(const h8*)(arow + n + half * 8);
    ((h8*)&a)[1] = *(const h8*)(arow + n + 16 + half * 8);
#pragma unroll
    for (int nt = 0; nt < 4; ++nt) {
      const f16* bp = k_t + tb + (size_t)(nt * 16 + lcol) * NTOK + n + half * 8;
      v16h bb;
      ((h8*)&bb)[0] = *(const h8*)bp;
      ((h8*)&bb)[1] = *(const h8*)(bp + 16);
      acc[nt] = WMMA_F16(a, bb, acc[nt]);
    }
  }
  float* out = SgramZ + ((size_t)z * 32 + bh) * (DD * DD);
#pragma unroll
  for (int nt = 0; nt < 4; ++nt)
#pragma unroll
    for (int j = 0; j < 8; ++j) {
      int d = wave * 16 + 8 * half + j;
      out[d * DD + nt * 16 + lcol] = acc[nt][j];
    }
}

// ---------------------------------------------------------------------------
// K3b: projection partials  out_z[b,h,d,kk] = sum_{n in slice} A[d,n] PT[kk,n]
// ---------------------------------------------------------------------------
__global__ __launch_bounds__(128) void k_projgemm(const f16* __restrict__ Amat,
                                                  const f16* __restrict__ PT,
                                                  float* __restrict__ outZ) {
  const int bh = blockIdx.x, z = blockIdx.y;
  const int wave = threadIdx.x >> 5, lane = threadIdx.x & 31;
  const int lcol = lane & 15, half = lane >> 4;
  v8f acc[16] = {};
  const f16* arow = Amat + (size_t)bh * DD * NTOK + (size_t)(wave * 16 + lcol) * NTOK;
  for (int n = z * (NTOK / NZ); n < (z + 1) * (NTOK / NZ); n += 32) {
    v16h a;
    ((h8*)&a)[0] = *(const h8*)(arow + n + half * 8);
    ((h8*)&a)[1] = *(const h8*)(arow + n + 16 + half * 8);
#pragma unroll
    for (int jt = 0; jt < 16; ++jt) {
      const f16* bp = PT + (size_t)(jt * 16 + lcol) * NTOK + n + half * 8;
      v16h bb;
      ((h8*)&bb)[0] = *(const h8*)bp;
      ((h8*)&bb)[1] = *(const h8*)(bp + 16);
      acc[jt] = WMMA_F16(a, bb, acc[jt]);
    }
  }
  float* out = outZ + ((size_t)z * 32 + bh) * (DD * KP);
#pragma unroll
  for (int jt = 0; jt < 16; ++jt)
#pragma unroll
    for (int j = 0; j < 8; ++j) {
      int d = wave * 16 + 8 * half + j;
      out[d * KP + jt * 16 + lcol] = acc[jt][j];
    }
}

// ---------------------------------------------------------------------------
// K4a: channel-attention softmax -> attn_ca f16 [bh][d][e]
// ---------------------------------------------------------------------------
__global__ void k_ca_softmax(const float* __restrict__ SgramZ,
                             const float* __restrict__ rq,
                             const float* __restrict__ rk,
                             const float* __restrict__ tca,
                             f16* __restrict__ attn_ca) {
  const int bh = blockIdx.x, d = threadIdx.x;
  const int h = bh & 7;
  float vals[64];
  float rqd = rq[bh * DD + d] * tca[h];
  float mx = -3.0e38f;
#pragma unroll
  for (int e = 0; e < 64; ++e) {
    float s = 0.f;
    for (int z = 0; z < NZ; ++z)
      s += SgramZ[((size_t)z * 32 + bh) * (DD * DD) + d * DD + e];
    s *= rqd * rk[bh * DD + e];
    vals[e] = s;
    mx = fmaxf(mx, s);
  }
  float sum = 0.f;
#pragma unroll
  for (int e = 0; e < 64; ++e) { float v = __expf(vals[e] - mx); vals[e] = v; sum += v; }
  float inv = 1.0f / sum;
#pragma unroll
  for (int e = 0; e < 64; ++e)
    attn_ca[(size_t)bh * (DD * DD) + d * DD + e] = (f16)(vals[e] * inv);
}

// ---------------------------------------------------------------------------
// K4b: finalize projections.
// ---------------------------------------------------------------------------
__global__ __launch_bounds__(256) void k_proj_fin(const float* __restrict__ kprojZ,
                                                  const float* __restrict__ vprojZ,
                                                  const float* __restrict__ rq,
                                                  const float* __restrict__ rk,
                                                  const float* __restrict__ tsa,
                                                  f16* __restrict__ kprojT,
                                                  f16* __restrict__ vprojS) {
  int idx = blockIdx.x * 256 + threadIdx.x;       // over 32*64*256
  int bh = idx >> 14, rem = idx & 16383;
  int d = rem >> 8, kk = rem & 255;
  float sk = 0.f, sv = 0.f;
  for (int z = 0; z < NZ; ++z) {
    size_t base = ((size_t)z * 32 + bh) * (DD * KP) + d * KP + kk;
    sk += kprojZ[base];
    sv += vprojZ[base];
  }
  float scale = rq[bh * DD + d] * rk[bh * DD + d] * tsa[bh & 7];
  kprojT[((size_t)bh * KP + kk) * DD + d] = (f16)(sk * scale);
  vprojS[((size_t)bh * DD + d) * KP + kk] = (f16)sv;
}

// ---------------------------------------------------------------------------
// K5: x_ca[d,n] = sum_e attn_ca[d,e] * vca[e,n]  -> WRITES d_out (f32)
// ---------------------------------------------------------------------------
__global__ __launch_bounds__(128) void k_xca(const f16* __restrict__ attn_ca,
                                             const f16* __restrict__ vca_nat,
                                             float* __restrict__ out) {
  const int bh = blockIdx.x;
  const int b = bh >> 3, h = bh & 7;
  const int wave = threadIdx.x >> 5, lane = threadIdx.x & 31;
  const int lcol = lane & 15, half = lane >> 4;
  const int nbase = blockIdx.y * 512 + wave * 128;
  const f16* A0 = attn_ca + (size_t)bh * (DD * DD);

  v16h a[4][2];
#pragma unroll
  for (int dt = 0; dt < 4; ++dt)
#pragma unroll
    for (int et = 0; et < 2; ++et) {
      const f16* p = A0 + (dt * 16 + lcol) * DD + et * 32 + half * 8;
      ((h8*)&a[dt][et])[0] = *(const h8*)p;
      ((h8*)&a[dt][et])[1] = *(const h8*)(p + 16);
    }
  for (int nt = 0; nt < 8; ++nt) {
    int n = nbase + nt * 16 + lcol;
    size_t rowoff = ((size_t)(b * NTOK + n)) * CCH + h * DD;
    v8f acc[4] = {};
#pragma unroll
    for (int et = 0; et < 2; ++et) {
      const f16* bp = vca_nat + rowoff + et * 32 + half * 8;
      v16h bb;
      ((h8*)&bb)[0] = *(const h8*)bp;
      ((h8*)&bb)[1] = *(const h8*)(bp + 16);
#pragma unroll
      for (int dt = 0; dt < 4; ++dt) acc[dt] = WMMA_F16(a[dt][et], bb, acc[dt]);
    }
#pragma unroll
    for (int dt = 0; dt < 4; ++dt)
      *(v8f*)(out + rowoff + dt * 16 + 8 * half) = acc[dt];
  }
}

// ---------------------------------------------------------------------------
// K6: fused spatial attention (16 token rows per wave)
// ---------------------------------------------------------------------------
__global__ __launch_bounds__(128) void k_spatial(const f16* __restrict__ q_nat,
                                                 const f16* __restrict__ kprojT,
                                                 const f16* __restrict__ vprojS,
                                                 float* __restrict__ out) {
  __shared__ f16 ldsbuf[4 * 4224];   // per wave: 16 rows x 264 f16
  const int bh = blockIdx.x;
  const int b = bh >> 3, h = bh & 7;
  const int wave = threadIdx.x >> 5, lane = threadIdx.x & 31;
  const int lcol = lane & 15, half = lane >> 4;
  const int n0 = blockIdx.y * 64 + wave * 16;
  f16* myl = ldsbuf + wave * 4224;

  size_t qrow = ((size_t)(b * NTOK + n0 + lcol)) * CCH + h * DD;
  v16h a[2];
#pragma unroll
  for (int kt = 0; kt < 2; ++kt) {
    const f16* p = q_nat + qrow + kt * 32 + half * 8;
    ((h8*)&a[kt])[0] = *(const h8*)p;
    ((h8*)&a[kt])[1] = *(const h8*)(p + 16);
  }
  v8f acc[16] = {};
  const f16* KPm = kprojT + (size_t)bh * (KP * DD);
#pragma unroll
  for (int jt = 0; jt < 16; ++jt) {
    const f16* col = KPm + (size_t)(jt * 16 + lcol) * DD;
#pragma unroll
    for (int kt = 0; kt < 2; ++kt) {
      const f16* bp = col + kt * 32 + half * 8;
      v16h bb;
      ((h8*)&bb)[0] = *(const h8*)bp;
      ((h8*)&bb)[1] = *(const h8*)(bp + 16);
      acc[jt] = WMMA_F16(a[kt], bb, acc[jt]);
    }
  }
  float rowsum[8];
#pragma unroll
  for (int j = 0; j < 8; ++j) {
    float m = acc[0][j];
#pragma unroll
    for (int jt = 1; jt < 16; ++jt) m = fmaxf(m, acc[jt][j]);
    m = fmaxf(m, __shfl_xor(m, 1, 32));
    m = fmaxf(m, __shfl_xor(m, 2, 32));
    m = fmaxf(m, __shfl_xor(m, 4, 32));
    m = fmaxf(m, __shfl_xor(m, 8, 32));
    float s = 0.f;
#pragma unroll
    for (int jt = 0; jt < 16; ++jt) {
      float e = __expf(acc[jt][j] - m);
      acc[jt][j] = e;
      s += e;
    }
    s += __shfl_xor(s, 1, 32);
    s += __shfl_xor(s, 2, 32);
    s += __shfl_xor(s, 4, 32);
    s += __shfl_xor(s, 8, 32);
    rowsum[j] = s;
  }
#pragma unroll
  for (int jt = 0; jt < 16; ++jt)
#pragma unroll
    for (int j = 0; j < 8; ++j)
      myl[(j + 8 * half) * 264 + jt * 16 + lcol] = (f16)acc[jt][j];

  v8f acc2[4] = {};
  const f16* VP = vprojS + (size_t)bh * (DD * KP);
#pragma unroll
  for (int kt = 0; kt < 8; ++kt) {
    const f16* ap = myl + lcol * 264 + kt * 32 + half * 8;
    v16h a2;
    ((h8*)&a2)[0] = *(const h8*)ap;
    ((h8*)&a2)[1] = *(const h8*)(ap + 16);
#pragma unroll
    for (int dt = 0; dt < 4; ++dt) {
      const f16* bp = VP + (size_t)(dt * 16 + lcol) * KP + kt * 32 + half * 8;
      v16h bb;
      ((h8*)&bb)[0] = *(const h8*)bp;
      ((h8*)&bb)[1] = *(const h8*)(bp + 16);
      acc2[dt] = WMMA_F16(a2, bb, acc2[dt]);
    }
  }
  __syncthreads();           // fence before retyping the LDS buffer as f32
  float* ol = (float*)myl;   // 16 rows x pitch 66 f32
#pragma unroll
  for (int j = 0; j < 8; ++j) {
    float rinv = 1.0f / rowsum[j];
#pragma unroll
    for (int dt = 0; dt < 4; ++dt)
      ol[(j + 8 * half) * 66 + dt * 16 + lcol] = acc2[dt][j] * rinv;
  }
#pragma unroll
  for (int r = 0; r < 16; ++r) {
    size_t o = ((size_t)(b * NTOK + n0 + r)) * CCH + h * DD + lane * 2;
    f2 v = *(const f2*)(ol + r * 66 + lane * 2);
    f2 g = *(f2*)(out + o);
    g.x += v.x;
    g.y += v.y;
    *(f2*)(out + o) = g;
  }
}

// ---------------------------------------------------------------------------
extern "C" void kernel_launch(void* const* d_in, const int* in_sizes, int n_in,
                              void* d_out, int out_size, void* d_ws, size_t ws_size,
                              hipStream_t stream) {
  (void)in_sizes; (void)n_in; (void)out_size; (void)ws_size;
  const float* X   = (const float*)d_in[0];
  const float* W   = (const float*)d_in[1];
  const float* P   = (const float*)d_in[2];
  const float* tca = (const float*)d_in[3];
  const float* tsa = (const float*)d_in[4];
  float* out = (float*)d_out;

  char* wsp = (char*)d_ws;
  size_t off = 0;
  auto take = [&](size_t bytes) -> void* {
    void* p = wsp + off;
    off += (bytes + 255) & ~(size_t)255;
    return p;
  };
  const size_t tsz = (size_t)ROWS * CCH * sizeof(f16);     // 32 MB per tensor
  f16* q_nat   = (f16*)take(tsz);
  f16* vca_nat = (f16*)take(tsz);
  f16* q_t     = (f16*)take(tsz);
  f16* k_t     = (f16*)take(tsz);
  f16* vsa_t   = (f16*)take(tsz);
  f16* X16     = (f16*)take(tsz);
  f16* WT16    = (f16*)take((size_t)C4 * CCH * sizeof(f16));
  f16* PT      = (f16*)take((size_t)KP * NTOK * sizeof(f16));
  float* rq    = (float*)take((size_t)BB * HH * DD * sizeof(float));
  float* rk    = (float*)take((size_t)BB * HH * DD * sizeof(float));
  float* SgramZ = (float*)take((size_t)NZ * 32 * DD * DD * sizeof(float));
  float* kprojZ = (float*)take((size_t)NZ * 32 * DD * KP * sizeof(float));
  float* vprojZ = (float*)take((size_t)NZ * 32 * DD * KP * sizeof(float));
  f16* attn_ca = (f16*)take((size_t)32 * DD * DD * sizeof(f16));
  f16* kprojT  = (f16*)take((size_t)32 * KP * DD * sizeof(f16));
  f16* vprojS  = (f16*)take((size_t)32 * DD * KP * sizeof(f16));

  k_cvt_x<<<(ROWS * CCH / 4) / 256, 256, 0, stream>>>(X, X16);
  k_transpose_w<<<dim3(CCH / 64, C4 / 64), 256, 0, stream>>>(W, WT16);
  k_transpose_P<<<dim3(NTOK / 64, KP / 64), 256, 0, stream>>>(P, PT);
  k_qkvv<<<dim3(ROWS / 128, C4 / 128), 256, 0, stream>>>(X16, WT16, q_nat, vca_nat, q_t, k_t, vsa_t);
  k_norms<<<BB * HH * DD, 256, 0, stream>>>(q_t, k_t, rq, rk);
  k_gram<<<dim3(32, NZ), 128, 0, stream>>>(q_t, k_t, SgramZ);
  k_projgemm<<<dim3(32, NZ), 128, 0, stream>>>(k_t, PT, kprojZ);
  k_projgemm<<<dim3(32, NZ), 128, 0, stream>>>(vsa_t, PT, vprojZ);
  k_ca_softmax<<<32, 64, 0, stream>>>(SgramZ, rq, rk, tca, attn_ca);
  k_proj_fin<<<2048, 256, 0, stream>>>(kprojZ, vprojZ, rq, rk, tsa, kprojT, vprojS);
  k_xca<<<dim3(32, 16), 128, 0, stream>>>(attn_ca, vca_nat, out);
  k_spatial<<<dim3(32, 128), 128, 0, stream>>>(q_nat, kprojT, vprojS, out);
}
